// AttentionalEmbed_62500364091665
// MI455X (gfx1250) — matrix operand order repaired and checked by reading, not compiled, not run on hardware
//
#include <hip/hip_runtime.h>
#include <hip/hip_bf16.h>

// Problem constants (from reference): Q=8192, G=8192, D=512, OUT=512
#define QN 8192
#define GN 8192
#define DN 512
#define ON 512

typedef __attribute__((ext_vector_type(16))) __bf16 v16bf;
typedef __attribute__((ext_vector_type(8)))  __bf16 v8bf;
typedef __attribute__((ext_vector_type(8)))  float  v8f;

// ---------------------------------------------------------------------------
// CDNA5 async global->LDS copy (ASYNCcnt-tracked, no VGPR data round trip)
// ---------------------------------------------------------------------------
__device__ __forceinline__ void async_b128(unsigned lds_off, const __bf16* g) {
    asm volatile("global_load_async_to_lds_b128 %0, %1, off"
                 :: "v"(lds_off), "v"(g) : "memory");
}
__device__ __forceinline__ void wait_async0() {
    asm volatile("s_wait_asynccnt 0" ::: "memory");
}

// ---------------------------------------------------------------------------
// fp32 -> bf16 convert with strided destination rows (lets us drop the
// query columns straight into the concatenated x=[q|r] buffer).
// ---------------------------------------------------------------------------
__global__ void cvt_rows_bf16(const float* __restrict__ src,
                              __bf16* __restrict__ dst,
                              long n, int cols, int dstld) {
    long i = (long)blockIdx.x * 256 + threadIdx.x;
    if (i < n) {
        long r = i / cols;
        int  c = (int)(i - r * cols);
        dst[r * (long)dstld + c] = (__bf16)src[i];
    }
}

// ---------------------------------------------------------------------------
// fp32 [rows, cols] -> bf16 transposed [cols, rows] (LDS tiled)
// ---------------------------------------------------------------------------
__global__ void transpose_bf16(const float* __restrict__ src,
                               __bf16* __restrict__ dst,
                               int rows, int cols) {
    __shared__ float t[32][33];
    int c0 = blockIdx.x * 32;
    int r0 = blockIdx.y * 32;
    int tx = threadIdx.x, ty = threadIdx.y;
#pragma unroll
    for (int i = 0; i < 4; ++i)
        t[ty + i * 8][tx] = src[(size_t)(r0 + ty + i * 8) * cols + c0 + tx];
    __syncthreads();
#pragma unroll
    for (int i = 0; i < 4; ++i)
        dst[(size_t)(c0 + ty + i * 8) * rows + r0 + tx] = (__bf16)t[tx][ty + i * 8];
}

// ---------------------------------------------------------------------------
// Row softmax: one workgroup per row, fp32 in -> bf16 probabilities out.
// ---------------------------------------------------------------------------
__global__ void softmax_rows(const float* __restrict__ S,
                             __bf16* __restrict__ P, int ncols) {
    __shared__ float red[256];
    int row = blockIdx.x;
    const float* s = S + (size_t)row * ncols;
    int tid = threadIdx.x;

    float mx = -3.4e38f;
    for (int c = tid; c < ncols; c += 256) mx = fmaxf(mx, s[c]);
    red[tid] = mx; __syncthreads();
    for (int off = 128; off; off >>= 1) {
        if (tid < off) red[tid] = fmaxf(red[tid], red[tid + off]);
        __syncthreads();
    }
    mx = red[0]; __syncthreads();

    float sum = 0.f;
    for (int c = tid; c < ncols; c += 256) sum += __expf(s[c] - mx);
    red[tid] = sum; __syncthreads();
    for (int off = 128; off; off >>= 1) {
        if (tid < off) red[tid] += red[tid + off];
        __syncthreads();
    }
    float inv = 1.f / red[0];

    __bf16* p = P + (size_t)row * ncols;
    for (int c = tid; c < ncols; c += 256)
        p[c] = (__bf16)(__expf(s[c] - mx) * inv);
}

// ---------------------------------------------------------------------------
// NT GEMM on WMMA: C[M,N] = A[M,K] * B[N,K]^T, bf16 inputs, fp32 accum.
// grid = (N/128, M/128), block = 256 threads (8 waves).
// WG tile 128x128; wave tile 32(M) x 64(N) = 2x4 WMMA 16x16 tiles; K-step 32.
// LDS double-buffered; tiles staged with GLOBAL_LOAD_ASYNC_TO_LDS_B128.
// MODE 0: store bf16, MODE 1: store f32, MODE 2: store f32 tanh(c + bias[n]).
// ---------------------------------------------------------------------------
#define LDSLD 40   // padded LDS row stride in bf16 elements (80 bytes)

template <int MODE>
__global__ __launch_bounds__(256)
void gemm_nt_wmma(const __bf16* __restrict__ A, int lda,
                  const __bf16* __restrict__ B, int ldb,
                  void* __restrict__ C, int ldc,
                  int K, const float* __restrict__ bias) {
    __shared__ __bf16 As[2][128 * LDSLD];
    __shared__ __bf16 Bs[2][128 * LDSLD];

    const int tid   = threadIdx.x;
    const int lane  = tid & 31;
    const int wid   = tid >> 5;
    const int waveM = wid & 3;   // 0..3 -> 4 x 32 rows
    const int waveN = wid >> 2;  // 0..1 -> 2 x 64 cols
    const int lr    = lane & 15;
    const int hs    = lane >> 4; // half-wave select

    const int bm = blockIdx.y * 128;
    const int bn = blockIdx.x * 128;

    // Per-thread staging slots: rows (tid>>2) and (tid>>2)+64, 16B column c0.
    const int row0 = tid >> 2;
    const int c0   = (tid & 3) * 8;
    const __bf16* ga0 = A + (size_t)(bm + row0) * lda + c0;
    const __bf16* ga1 = ga0 + (size_t)64 * lda;
    const __bf16* gb0 = B + (size_t)(bn + row0) * ldb + c0;
    const __bf16* gb1 = gb0 + (size_t)64 * ldb;

    const unsigned BUFB = 128 * LDSLD * 2;  // bytes per LDS buffer
    const unsigned HALF = 64 * LDSLD * 2;   // 64-row offset in bytes
    const unsigned laBase = (unsigned)(size_t)(&As[0][0]) + (unsigned)(row0 * LDSLD + c0) * 2;
    const unsigned lbBase = (unsigned)(size_t)(&Bs[0][0]) + (unsigned)(row0 * LDSLD + c0) * 2;

    v8f acc[2][4] = {};

    // Prologue: stage K-block 0 into buffer 0.
    async_b128(laBase,        ga0);
    async_b128(laBase + HALF, ga1);
    async_b128(lbBase,        gb0);
    async_b128(lbBase + HALF, gb1);
    wait_async0();
    __syncthreads();

    const int nk = K >> 5;
    for (int kb = 0; kb < nk; ++kb) {
        const int cur = kb & 1;

        // Prefetch next K-block into the alternate buffer (async, no wait).
        if (kb + 1 < nk) {
            const size_t  ko  = (size_t)(kb + 1) * 32;
            const unsigned db = (unsigned)(cur ^ 1) * BUFB;
            async_b128(laBase + db,        ga0 + ko);
            async_b128(laBase + db + HALF, ga1 + ko);
            async_b128(lbBase + db,        gb0 + ko);
            async_b128(lbBase + db + HALF, gb1 + ko);
        }

        const __bf16* as = &As[cur][0];
        const __bf16* bs = &Bs[cur][0];

        // ---- A fragments (16x32, ISA layout) ----
        v16bf afrag[2];
#pragma unroll
        for (int mt = 0; mt < 2; ++mt) {
            int arow = (waveM * 32 + mt * 16 + lr) * LDSLD;
            int kbse = hs * 8;  // lanes 0-15: K 0..7 & 16..23; 16-31: 8..15 & 24..31
            v8bf lo = *(const v8bf*)(&as[arow + kbse]);
            v8bf hi = *(const v8bf*)(&as[arow + kbse + 16]);
            afrag[mt] = __builtin_shufflevector(lo, hi,
                0,1,2,3,4,5,6,7,8,9,10,11,12,13,14,15);
        }
        // ---- B fragments (32x16, ISA layout, same K order as A) ----
        v16bf bfrag[4];
#pragma unroll
        for (int nt = 0; nt < 4; ++nt) {
            int brow = (waveN * 64 + nt * 16 + lr) * LDSLD;
            int kbse = hs * 8;  // lanes 0-15: K 0..7 & 16..23; 16-31: 8..15 & 24..31
            v8bf lo = *(const v8bf*)(&bs[brow + kbse]);
            v8bf hi = *(const v8bf*)(&bs[brow + kbse + 16]);
            bfrag[nt] = __builtin_shufflevector(lo, hi,
                0,1,2,3,4,5,6,7,8,9,10,11,12,13,14,15);
        }

#pragma unroll
        for (int mt = 0; mt < 2; ++mt)
#pragma unroll
            for (int nt = 0; nt < 4; ++nt)
                acc[mt][nt] = __builtin_amdgcn_wmma_f32_16x16x32_bf16(
                    false, afrag[mt], false, bfrag[nt],
                    (short)0, acc[mt][nt], false, false);

        // My async copies into the other buffer must have landed, and every
        // wave must be done reading `cur`, before the next iteration.
        wait_async0();
        __syncthreads();
    }

    // ---- epilogue: C/D layout -> lane holds (M = r + 8*hs, N = lr) ----
#pragma unroll
    for (int mt = 0; mt < 2; ++mt) {
#pragma unroll
        for (int nt = 0; nt < 4; ++nt) {
            int n = bn + waveN * 64 + nt * 16 + lr;
            int mbase = bm + waveM * 32 + mt * 16 + hs * 8;
#pragma unroll
            for (int r = 0; r < 8; ++r) {
                float v = acc[mt][nt][r];
                size_t idx = (size_t)(mbase + r) * ldc + n;
                if (MODE == 0) {
                    ((__bf16*)C)[idx] = (__bf16)v;
                } else if (MODE == 1) {
                    ((float*)C)[idx] = v;
                } else {
                    ((float*)C)[idx] = tanhf(v + bias[n]);
                }
            }
        }
    }
}

// ---------------------------------------------------------------------------
// Host orchestration
// ---------------------------------------------------------------------------
extern "C" void kernel_launch(void* const* d_in, const int* in_sizes, int n_in,
                              void* d_out, int out_size, void* d_ws, size_t ws_size,
                              hipStream_t stream) {
    const float* query   = (const float*)d_in[0]; // [Q, D]
    const float* gallery = (const float*)d_in[1]; // [G, D]
    const float* label   = (const float*)d_in[2]; // [G, G]
    const float* W       = (const float*)d_in[3]; // [OUT, 2D]
    const float* bias    = (const float*)d_in[4]; // [OUT]

    char* ws = (char*)d_ws;
    size_t off = 0;
    __bf16* xbf      = (__bf16*)(ws + off); off += (size_t)QN * 2 * DN * 2; // [Q, 2D] = [q | r]
    __bf16* gbf      = (__bf16*)(ws + off); off += (size_t)GN * DN * 2;     // [G, D]
    __bf16* gbfT     = (__bf16*)(ws + off); off += (size_t)DN * GN * 2;     // [D, G]
    __bf16* wbf      = (__bf16*)(ws + off); off += (size_t)ON * 2 * DN * 2; // [OUT, 2D]
    __bf16* labelT   = (__bf16*)(ws + off); off += (size_t)GN * GN * 2;     // [G, G] transposed
    __bf16* scoresbf = (__bf16*)(ws + off); off += (size_t)QN * GN * 2;     // scores, reused as attn
    float*  sf32     = (float*) (ws + off); off += (size_t)QN * GN * 4;     // pre-softmax logits

    // --- precision staging: fp32 -> bf16 (+ transposes for the NT GEMMs) ---
    {
        long n = (long)QN * DN;
        cvt_rows_bf16<<<(n + 255) / 256, 256, 0, stream>>>(query, xbf, n, DN, 2 * DN);
    }
    {
        long n = (long)GN * DN;
        cvt_rows_bf16<<<(n + 255) / 256, 256, 0, stream>>>(gallery, gbf, n, DN, DN);
    }
    {
        long n = (long)ON * 2 * DN;
        cvt_rows_bf16<<<(n + 255) / 256, 256, 0, stream>>>(W, wbf, n, 2 * DN, 2 * DN);
    }
    transpose_bf16<<<dim3(GN / 32, GN / 32), dim3(32, 8), 0, stream>>>(label, labelT, GN, GN);
    transpose_bf16<<<dim3(DN / 32, GN / 32), dim3(32, 8), 0, stream>>>(gallery, gbfT, GN, DN);

    // --- scores = Q . G^T  -> bf16 [Q, G] ---
    gemm_nt_wmma<0><<<dim3(GN / 128, QN / 128), 256, 0, stream>>>(
        xbf, 2 * DN, gbf, DN, scoresbf, GN, DN, nullptr);

    // --- s = scores . Y  (the 8192^3 GEMM) -> fp32 [Q, G] ---
    gemm_nt_wmma<1><<<dim3(GN / 128, QN / 128), 256, 0, stream>>>(
        scoresbf, GN, labelT, GN, sf32, GN, GN, nullptr);

    // --- attn = softmax(s) -> bf16, reuse scores buffer ---
    softmax_rows<<<QN, 256, 0, stream>>>(sf32, scoresbf, GN);

    // --- r = attn . G -> bf16, written into x's second half (ldc = 2D) ---
    gemm_nt_wmma<0><<<dim3(DN / 128, QN / 128), 256, 0, stream>>>(
        scoresbf, GN, gbfT, GN, xbf + DN, 2 * DN, GN, nullptr);

    // --- out = tanh(x . W^T + b) -> fp32 [Q, OUT] ---
    gemm_nt_wmma<2><<<dim3(ON / 128, QN / 128), 256, 0, stream>>>(
        xbf, 2 * DN, wbf, 2 * DN, (float*)d_out, ON, 2 * DN, bias);
}
